// GraphNetBlock_54322746359851
// MI455X (gfx1250) — matrix operand order repaired and checked by reading, not compiled, not run on hardware
//
#include <hip/hip_runtime.h>
#include <hip/hip_bf16.h>

typedef __attribute__((ext_vector_type(16))) _Float16 v16h;
typedef __attribute__((ext_vector_type(8)))  float    v8f;

#define H 128
#define TILE_M 32          // rows (edges/nodes) per workgroup
#define NT 8               // 128/16 output column tiles

// Native non-returning fp32 global atomic add (CDNA5 VGLOBAL op 86,
// TH[0]=0 -> no return, tracked with STOREcnt). Avoids any CAS fallback.
__device__ inline void atomic_add_f32(float* p, float v) {
    asm volatile("global_atomic_add_f32 %0, %1, off"
                 :
                 : "v"((unsigned long long)(__builtin_bit_cast(unsigned long long, p))),
                   "v"(v)
                 : "memory");
}

// ---------------------------------------------------------------------------
// Weight repack: fp32 row-major W[K][128] -> f16 WMMA B-fragments.
// Fragment layout: frag[((kt*8 + nt)*32 + lane)*16 + j]  with
//   K = kt*32 + (j>>3)*16 + (lane>=16 ? 8 : 0) + (j&7) ,  N = nt*16 + (lane & 15)
// (ISA 16-bit B 32x16: lanes 0-15 K=0..7,16..23, lanes 16-31 K=8..15,24..31,
//  VGPR v holds K=(v>>2)*16+(v&3)*2+{0,1}, +8 in lanes 16-31.)
// ---------------------------------------------------------------------------
__global__ void convert_w(const float* __restrict__ W, _Float16* __restrict__ dst, int K) {
    int idx = blockIdx.x * blockDim.x + threadIdx.x;
    int total = K * H;
    if (idx >= total) return;
    int j    = idx & 15;
    int lane = (idx >> 4) & 31;
    int nt   = (idx >> 9) & 7;
    int kt   = idx >> 12;
    int k = kt * 32 + ((j >> 3) << 4) + ((lane >> 4) << 3) + (j & 7);
    int n = nt * 16 + (lane & 15);
    dst[idx] = (_Float16)W[(size_t)k * H + n];
}

// ---------------------------------------------------------------------------
// One 16(M)x64(N) wave tile of a GEMM layer: acc[4] over 4 N-tiles,
// accumulating over KT K-tiles of 32. Bias folded into accumulator init
// (bias depends only on N == lane&15 -> same value in all 8 M-slots).
// A-frag from LDS per ISA 16-bit A 16x32 layout:
//   element 2v+j : K = (v>>2)*16 + (lane>>4)*8 + (v&3)*2 + j , M = lane&15
// (contiguous per 8 halfs -> compiler emits 2x ds_load_b128 per frag)
// ---------------------------------------------------------------------------
template <int KT>
__device__ inline void gemm4(const _Float16* __restrict__ actIn, int stride,
                             const _Float16* __restrict__ wfrag,
                             const float* __restrict__ bias,
                             int lane, int mtile, int nbase, v8f acc[4]) {
    const int col = lane & 15;
    const int hi  = lane >> 4;
#pragma unroll
    for (int nn = 0; nn < 4; ++nn) {
        float bv = bias[(nbase + nn) * 16 + col];
        v8f c;
#pragma unroll
        for (int v = 0; v < 8; ++v) c[v] = bv;
        acc[nn] = c;
    }
#pragma unroll
    for (int kt = 0; kt < KT; ++kt) {
        const _Float16* pa = actIn + (mtile * 16 + col) * stride + kt * 32;
        v16h a;
#pragma unroll
        for (int v = 0; v < 8; ++v) {
            int K = ((v & 4) << 2) + (hi << 3) + ((v & 3) << 1);
            a[2 * v]     = pa[K];
            a[2 * v + 1] = pa[K + 1];
        }
#pragma unroll
        for (int nn = 0; nn < 4; ++nn) {
            const v16h b = *(const v16h*)(wfrag + (((size_t)(kt * NT + nbase + nn)) * 32 + lane) * 16);
            acc[nn] = __builtin_amdgcn_wmma_f32_16x16x32_f16(
                false, a, false, b, (short)0, acc[nn], false, false);
        }
    }
}

__device__ inline void store_relu_f16(_Float16* __restrict__ out, int stride,
                                      int lane, int mtile, int nbase, const v8f acc[4]) {
    const int col = lane & 15;
    const int hi  = lane >> 4;
#pragma unroll
    for (int nn = 0; nn < 4; ++nn)
#pragma unroll
        for (int v = 0; v < 8; ++v) {
            int row = mtile * 16 + hi * 8 + v;
            int c   = (nbase + nn) * 16 + col;
            float x = acc[nn][v];
            out[row * stride + c] = (_Float16)(x > 0.f ? x : 0.f);
        }
}

__device__ inline void store_f32(float* __restrict__ out, int stride,
                                 int lane, int mtile, int nbase, const v8f acc[4]) {
    const int col = lane & 15;
    const int hi  = lane >> 4;
#pragma unroll
    for (int nn = 0; nn < 4; ++nn)
#pragma unroll
        for (int v = 0; v < 8; ++v)
            out[(mtile * 16 + hi * 8 + v) * stride + (nbase + nn) * 16 + col] = acc[nn][v];
}

// ---------------------------------------------------------------------------
// Edge kernel: 32 edges per block, 128 threads (4 waves).
//   input  = [edge_attr | x_i | rope(x_j)]  (384 cols, f16 in LDS)
//   4-layer MLP -> RMSNorm -> edge_out = edge_attr + upd ; aggr += upd (atomic)
// wave w: M-tile = w&1 (16 rows), N-half = w>>1 (4 of 8 N-tiles)
// ---------------------------------------------------------------------------
#define SIN_E 392   // 384 + pad
#define SACT  136   // 128 + pad
#define SOUT  132   // 128 + pad (f32)

// -log2(10000)/21 for inv_freq = 2^(p * ROPE_NEG)
#define ROPE_NEG (-0.632799451915162f)

__global__ __launch_bounds__(128) void edge_kernel(
    const float* __restrict__ x, const float* __restrict__ edge_attr,
    const float* __restrict__ pos, const _Float16* __restrict__ wfrag,
    const float* __restrict__ b0, const float* __restrict__ b1,
    const float* __restrict__ b2, const float* __restrict__ b3,
    const float* __restrict__ scale,
    const int* __restrict__ senders, const int* __restrict__ receivers,
    float* __restrict__ edge_out, float* __restrict__ aggr, int E) {

    __shared__ _Float16 sIn[TILE_M * SIN_E];
    __shared__ _Float16 sA[TILE_M * SACT];
    __shared__ _Float16 sB[TILE_M * SACT];
    __shared__ float    sOut[TILE_M * SOUT];
    __shared__ float    sRms[TILE_M];

    const int t    = threadIdx.x;
    const int lane = t & 31;
    const int wid  = t >> 5;
    const int ebase = blockIdx.x * TILE_M;

    // ---- build input tile: 4 threads per edge, thread q handles 32 cols ----
    const int el = t >> 2;
    const int q  = t & 3;
    const int ec = min(ebase + el, E - 1);
    const int snd = senders[ec];
    const int rcv = receivers[ec];
    const float* xr = x + (size_t)rcv * H;
    const float* xs = x + (size_t)snd * H;

#pragma unroll 8
    for (int i = 0; i < 32; ++i) {
        int c = q * 32 + i;
        sIn[el * SIN_E + c]     = (_Float16)edge_attr[(size_t)ec * H + c];
        sIn[el * SIN_E + H + c] = (_Float16)xr[c];
    }
    float dp0 = pos[snd * 3 + 0] - pos[rcv * 3 + 0];
    float dp1 = pos[snd * 3 + 1] - pos[rcv * 3 + 1];
    float dp2 = pos[snd * 3 + 2] - pos[rcv * 3 + 2];
#pragma unroll 4
    for (int pi = 0; pi < 16; ++pi) {
        int pp = q * 16 + pi;                     // pair index: 63 rotated + 1 pass
        if (pp < 63) {
            int a = pp / 21, p = pp - a * 21;
            float dpa = (a == 0) ? dp0 : (a == 1 ? dp1 : dp2);
            float inv = exp2f((float)p * ROPE_NEG);
            float sn, cs;
            __sincosf(dpa * inv, &sn, &cs);
            float ev = xs[2 * pp], od = xs[2 * pp + 1];
            sIn[el * SIN_E + 2 * H + 2 * pp]     = (_Float16)(ev * cs - od * sn);
            sIn[el * SIN_E + 2 * H + 2 * pp + 1] = (_Float16)(ev * sn + od * cs);
        } else {
            sIn[el * SIN_E + 2 * H + 126] = (_Float16)xs[126];
            sIn[el * SIN_E + 2 * H + 127] = (_Float16)xs[127];
        }
    }
    __syncthreads();

    // ---- 4-layer MLP via WMMA ----
    const int mtile = wid & 1;
    const int nbase = (wid >> 1) * 4;
    v8f acc[4];

    gemm4<12>(sIn, SIN_E, wfrag,              b0, lane, mtile, nbase, acc);
    store_relu_f16(sA, SACT, lane, mtile, nbase, acc);
    __syncthreads();
    gemm4<4>(sA, SACT, wfrag + 49152,         b1, lane, mtile, nbase, acc);
    store_relu_f16(sB, SACT, lane, mtile, nbase, acc);
    __syncthreads();
    gemm4<4>(sB, SACT, wfrag + 49152 + 16384, b2, lane, mtile, nbase, acc);
    store_relu_f16(sA, SACT, lane, mtile, nbase, acc);
    __syncthreads();
    gemm4<4>(sA, SACT, wfrag + 49152 + 32768, b3, lane, mtile, nbase, acc);
    store_f32(sOut, SOUT, lane, mtile, nbase, acc);
    __syncthreads();

    // ---- RMSNorm factor: one thread per row ----
    if (t < TILE_M) {
        float ss = 0.f;
#pragma unroll 8
        for (int c = 0; c < H; ++c) { float v = sOut[t * SOUT + c]; ss += v * v; }
        sRms[t] = rsqrtf(ss * (1.f / H) + 1e-6f);
    }
    __syncthreads();

    // ---- write edge output + scatter-add into aggr ----
    const int eo = ebase + el;
    if (eo < E) {
        float rs = sRms[el];
#pragma unroll 8
        for (int i = 0; i < 32; ++i) {
            int c = q * 32 + i;
            float upd = sOut[el * SOUT + c] * rs * (1.f + scale[c]);
            edge_out[(size_t)eo * H + c] = edge_attr[(size_t)eo * H + c] + upd;
            atomic_add_f32(&aggr[(size_t)rcv * H + c], upd);
        }
    }
}

// ---------------------------------------------------------------------------
// Node kernel: 32 nodes per block; input = [x | aggr] (256 cols)
// ---------------------------------------------------------------------------
#define SIN_N 264   // 256 + pad

__global__ __launch_bounds__(128) void node_kernel(
    const float* __restrict__ x, const float* __restrict__ aggr,
    const _Float16* __restrict__ wfrag,
    const float* __restrict__ b0, const float* __restrict__ b1,
    const float* __restrict__ b2, const float* __restrict__ b3,
    const float* __restrict__ scale, float* __restrict__ x_out, int N) {

    __shared__ _Float16 sIn[TILE_M * SIN_N];
    __shared__ _Float16 sA[TILE_M * SACT];
    __shared__ _Float16 sB[TILE_M * SACT];
    __shared__ float    sOut[TILE_M * SOUT];
    __shared__ float    sRms[TILE_M];

    const int t    = threadIdx.x;
    const int lane = t & 31;
    const int wid  = t >> 5;
    const int nbase0 = blockIdx.x * TILE_M;

    const int nl = t >> 2;
    const int q  = t & 3;
    const int nc = min(nbase0 + nl, N - 1);
#pragma unroll 8
    for (int i = 0; i < 64; ++i) {
        int c = q * 64 + i;
        float v = (c < H) ? x[(size_t)nc * H + c] : aggr[(size_t)nc * H + (c - H)];
        sIn[nl * SIN_N + c] = (_Float16)v;
    }
    __syncthreads();

    const int mtile = wid & 1;
    const int nb = (wid >> 1) * 4;
    v8f acc[4];

    gemm4<8>(sIn, SIN_N, wfrag,               b0, lane, mtile, nb, acc);
    store_relu_f16(sA, SACT, lane, mtile, nb, acc);
    __syncthreads();
    gemm4<4>(sA, SACT, wfrag + 32768,         b1, lane, mtile, nb, acc);
    store_relu_f16(sB, SACT, lane, mtile, nb, acc);
    __syncthreads();
    gemm4<4>(sB, SACT, wfrag + 32768 + 16384, b2, lane, mtile, nb, acc);
    store_relu_f16(sA, SACT, lane, mtile, nb, acc);
    __syncthreads();
    gemm4<4>(sA, SACT, wfrag + 32768 + 32768, b3, lane, mtile, nb, acc);
    store_f32(sOut, SOUT, lane, mtile, nb, acc);
    __syncthreads();

    if (t < TILE_M) {
        float ss = 0.f;
#pragma unroll 8
        for (int c = 0; c < H; ++c) { float v = sOut[t * SOUT + c]; ss += v * v; }
        sRms[t] = rsqrtf(ss * (1.f / H) + 1e-6f);
    }
    __syncthreads();

    const int no = nbase0 + nl;
    if (no < N) {
        float rs = sRms[nl];
#pragma unroll 8
        for (int i = 0; i < 32; ++i) {
            int c = q * 32 + i;
            float upd = sOut[nl * SOUT + c] * rs * (1.f + scale[c]);
            x_out[(size_t)no * H + c] = x[(size_t)no * H + c] + upd;
        }
    }
}

extern "C" void kernel_launch(void* const* d_in, const int* in_sizes, int n_in,
                              void* d_out, int out_size, void* d_ws, size_t ws_size,
                              hipStream_t stream) {
    const float* x         = (const float*)d_in[0];
    const float* edge_attr = (const float*)d_in[1];
    const float* pos       = (const float*)d_in[2];
    const float* eW[4] = {(const float*)d_in[3], (const float*)d_in[4],
                          (const float*)d_in[5], (const float*)d_in[6]};
    const float* eB[4] = {(const float*)d_in[7], (const float*)d_in[8],
                          (const float*)d_in[9], (const float*)d_in[10]};
    const float* eScale = (const float*)d_in[11];
    const float* nW[4] = {(const float*)d_in[12], (const float*)d_in[13],
                          (const float*)d_in[14], (const float*)d_in[15]};
    const float* nB[4] = {(const float*)d_in[16], (const float*)d_in[17],
                          (const float*)d_in[18], (const float*)d_in[19]};
    const float* nScale = (const float*)d_in[20];
    const int* senders   = (const int*)d_in[21];
    const int* receivers = (const int*)d_in[22];

    const int E = in_sizes[21];
    const int N = in_sizes[0] / H;

    float* out_x = (float*)d_out;
    float* out_e = out_x + (size_t)N * H;

    char* ws = (char*)d_ws;
    float*     aggr  = (float*)ws;
    size_t     aggrB = (size_t)N * H * sizeof(float);
    _Float16*  ewf = (_Float16*)(ws + aggrB);           // 98304 halfs
    _Float16*  nwf = ewf + 98304;                       // 81920 halfs

    // repack weights to WMMA fragments (cheap, deterministic each call)
    auto cvt = [&](const float* W, _Float16* dst, int K) {
        int total = K * H;
        convert_w<<<(total + 255) / 256, 256, 0, stream>>>(W, dst, K);
    };
    cvt(eW[0], ewf,                 3 * H);
    cvt(eW[1], ewf + 49152,         H);
    cvt(eW[2], ewf + 49152 + 16384, H);
    cvt(eW[3], ewf + 49152 + 32768, H);
    cvt(nW[0], nwf,                 2 * H);
    cvt(nW[1], nwf + 32768,         H);
    cvt(nW[2], nwf + 32768 + 16384, H);
    cvt(nW[3], nwf + 32768 + 32768, H);

    (void)hipMemsetAsync(aggr, 0, aggrB, stream);

    edge_kernel<<<(E + TILE_M - 1) / TILE_M, 128, 0, stream>>>(
        x, edge_attr, pos, ewf, eB[0], eB[1], eB[2], eB[3], eScale,
        senders, receivers, out_e, aggr, E);

    node_kernel<<<(N + TILE_M - 1) / TILE_M, 128, 0, stream>>>(
        x, aggr, nwf, nB[0], nB[1], nB[2], nB[3], nScale, out_x, N);
}
